// AttentionHead_13649406066726
// MI455X (gfx1250) — matrix operand order repaired and checked by reading, not compiled, not run on hardware
//
#include <hip/hip_runtime.h>

// Problem constants (from reference): B=8, T=4096, C=1024, H=128
#define B_ 8
#define T_ 4096
#define C_ 1024
#define H_ 128

typedef __attribute__((ext_vector_type(16))) __bf16 v16bf;
typedef __attribute__((ext_vector_type(8)))  __bf16 v8bf;
typedef __attribute__((ext_vector_type(2)))  __bf16 v2bf;
typedef __attribute__((ext_vector_type(8)))  float  v8f;

// f32 -> bf16 through the compiler's native cast (lets the backend pick
// v_cvt_pk_bf16_f32 / v_cvt_bf16_f32 on gfx1250 instead of integer RNE math).
__device__ __forceinline__ unsigned short f2bf(float f) {
  __bf16 h = (__bf16)f;
  return __builtin_bit_cast(unsigned short, h);
}

// Packed f32x2 -> bf16x2.
__device__ __forceinline__ unsigned int f2bf_pk(float a, float b) {
#if __has_builtin(__builtin_amdgcn_cvt_pk_bf16_f32)
  v2bf p = __builtin_amdgcn_cvt_pk_bf16_f32(a, b);
  return __builtin_bit_cast(unsigned int, p);
#else
  v2bf p;
  p[0] = (__bf16)a;
  p[1] = (__bf16)b;
  return __builtin_bit_cast(unsigned int, p);
#endif
}

__device__ __forceinline__ float fexp2(float x) {
#if __has_builtin(__builtin_amdgcn_exp2f)
  return __builtin_amdgcn_exp2f(x);       // raw v_exp_f32
#else
  return exp2f(x);
#endif
}

// ---------------------------------------------------------------------------
// Kernel 0: W [C][H] f32  ->  Wt [3][H][C] bf16 (transposed so that WMMA
// B-fragment loads along K(C) are contiguous 32-byte loads per lane).
// ---------------------------------------------------------------------------
__global__ void wt_kernel(const float* __restrict__ Wq, const float* __restrict__ Wk,
                          const float* __restrict__ Wv, unsigned short* __restrict__ Wt) {
  int idx = blockIdx.x * 256 + threadIdx.x;
  if (idx >= 3 * H_ * C_) return;
  int m = idx / (H_ * C_);
  int r = idx - m * (H_ * C_);
  int h = r / C_;
  int c = r - h * C_;
  const float* W = (m == 0) ? Wq : ((m == 1) ? Wk : Wv);
  Wt[idx] = f2bf(W[(size_t)c * H_ + h]);
}

// ---------------------------------------------------------------------------
// Kernel 1: QKV projection.  One wave computes a 16-row x 128-col tile of one
// of {Q, K, V} via 32 K-steps of v_wmma_f32_16x16x32_bf16 (8 N-tiles each).
//   Q  -> Qb [B*T][H] bf16, pre-scaled by log2(e)/sqrt(H)  (exp2-domain softmax)
//   K  -> Kb [B*T][H] bf16
//   V  -> Vt [B][H][T] bf16 (transposed for contiguous B-frag loads in attn)
// ---------------------------------------------------------------------------
__global__ __launch_bounds__(256) void qkv_proj(
    const float* __restrict__ x,
    const float* __restrict__ bq, const float* __restrict__ bk, const float* __restrict__ bv,
    const unsigned short* __restrict__ Wt,
    unsigned short* __restrict__ Qb, unsigned short* __restrict__ Kb,
    unsigned short* __restrict__ Vt)
{
  const int lane = threadIdx.x & 31;
  const int w    = threadIdx.x >> 5;
  const int col  = lane & 15;
  const int half = lane >> 4;
  const int m    = blockIdx.y;                 // 0=Q 1=K 2=V
  const int tile = blockIdx.x * 8 + w;         // 16-row tile index in [0, B*T/16)
  const int t0   = tile * 16;
  const unsigned short* Wm = Wt + (size_t)m * H_ * C_;

  v8f zero = {0.f,0.f,0.f,0.f,0.f,0.f,0.f,0.f};
  v8f acc[8];
  #pragma unroll
  for (int i = 0; i < 8; ++i) acc[i] = zero;

  for (int kb = 0; kb < C_; kb += 32) {
    // A fragment: 16x32 bf16 of x.  lane<16: row=lane, k 0..15;
    // lane>=16: row=lane-16, k 16..31.
    v16bf a;
    {
      const float* p = x + (size_t)(t0 + col) * C_ + kb + half * 16;
      float4 f0 = *(const float4*)(p);
      float4 f1 = *(const float4*)(p + 4);
      float4 f2 = *(const float4*)(p + 8);
      float4 f3 = *(const float4*)(p + 12);
      union { unsigned int u[8]; v16bf v; } af;
      af.u[0] = f2bf_pk(f0.x, f0.y);
      af.u[1] = f2bf_pk(f0.z, f0.w);
      af.u[2] = f2bf_pk(f1.x, f1.y);
      af.u[3] = f2bf_pk(f1.z, f1.w);
      af.u[4] = f2bf_pk(f2.x, f2.y);
      af.u[5] = f2bf_pk(f2.z, f2.w);
      af.u[6] = f2bf_pk(f3.x, f3.y);
      af.u[7] = f2bf_pk(f3.z, f3.w);
      a = af.v;
    }
    #pragma unroll
    for (int nt = 0; nt < 8; ++nt) {
      // B fragment: 32x16 bf16 of W^T. lane<16: col=lane, k 0..15; lane>=16: k 16..31.
      v16bf bfr = *(const v16bf*)(Wm + (size_t)(nt * 16 + col) * C_ + kb + half * 16);
      acc[nt] = __builtin_amdgcn_wmma_f32_16x16x32_bf16(
          false, a, false, bfr, (short)0, acc[nt], false, false);
    }
  }

  const float* bias = (m == 0) ? bq : ((m == 1) ? bk : bv);
  if (m == 2) {
    const int b  = t0 / T_;
    const int tl = (t0 % T_) + half * 8;       // 8 consecutive t per lane -> 16B packed store
    #pragma unroll
    for (int nt = 0; nt < 8; ++nt) {
      int h = nt * 16 + col;
      float bb = bias[h];
      union { unsigned int u[4]; uint4 v; } pk;
      #pragma unroll
      for (int r = 0; r < 4; ++r)
        pk.u[r] = f2bf_pk(acc[nt][2 * r] + bb, acc[nt][2 * r + 1] + bb);
      *(uint4*)(Vt + ((size_t)b * H_ + h) * T_ + tl) = pk.v;
    }
  } else {
    unsigned short* O = (m == 0) ? Qb : Kb;
    // log2(e)/sqrt(128) folded into Q so attention can use raw v_exp_f32 (base 2)
    const float s = (m == 0) ? (1.4426950408889634f * 0.08838834764831845f) : 1.0f;
    #pragma unroll
    for (int nt = 0; nt < 8; ++nt) {
      int h = nt * 16 + col;
      float bb = bias[h];
      #pragma unroll
      for (int r = 0; r < 8; ++r) {
        int t = t0 + r + half * 8;
        O[(size_t)t * H_ + h] = f2bf((acc[nt][r] + bb) * s);
      }
    }
  }
}

// ---------------------------------------------------------------------------
// Kernel 2: causal flash attention.  One wave owns 16 query rows x H=128.
// Key blocks of 64: S(16x64) via 16 WMMAs, online softmax in exp2 domain
// (shfl_xor within 16-lane halves + raw v_exp_f32), P -> bf16 A-layout via
// per-wave LDS slice, O += P@V via 16 WMMAs on transposed V.
// ---------------------------------------------------------------------------
__global__ __launch_bounds__(256) void flash_attn(
    const unsigned short* __restrict__ Qb, const unsigned short* __restrict__ Kb,
    const unsigned short* __restrict__ Vt, float* __restrict__ out)
{
  __shared__ unsigned short lds[8][16 * 64];   // 2KB per wave (16x64 P tile)
  const int lane = threadIdx.x & 31;
  const int w    = threadIdx.x >> 5;
  const int col  = lane & 15;
  const int half = lane >> 4;
  const int task = blockIdx.x * 8 + w;         // in [0, B * T/16)
  const int b    = task >> 8;                  // T/16 = 256 tiles per batch
  const int qt   = task & 255;
  const int q0   = qt * 16;

  // Q fragments (4 chunks of K=32), held in registers for the whole pass.
  v16bf qf[4];
  {
    const unsigned short* qp = Qb + (size_t)(b * T_ + q0 + col) * H_;
    #pragma unroll
    for (int hc = 0; hc < 4; ++hc) {
      int k0 = hc * 32 + half * 16;
      v8bf lo = *(const v8bf*)(qp + k0);
      v8bf hi = *(const v8bf*)(qp + k0 + 8);
      qf[hc] = __builtin_shufflevector(lo, hi, 0,1,2,3,4,5,6,7,8,9,10,11,12,13,14,15);
    }
  }

  v8f zero = {0.f,0.f,0.f,0.f,0.f,0.f,0.f,0.f};
  v8f o[8];
  #pragma unroll
  for (int i = 0; i < 8; ++i) o[i] = zero;
  float mrow[8], lrow[8];
  #pragma unroll
  for (int r = 0; r < 8; ++r) { mrow[r] = -3.0e38f; lrow[r] = 0.f; }

  const int nblocks = (q0 + 79) >> 6;          // keys 0 .. q0+15, blocks of 64
  for (int kb = 0; kb < nblocks; ++kb) {
    const int kbase = kb * 64;

    if (kb + 1 < nblocks)                      // gfx1250 global_prefetch_b8
      __builtin_prefetch(Kb + (size_t)(b * T_ + kbase + 64 + col) * H_, 0, 3);

    // S = Q @ K^T for this 64-key block (four 16x16 C-fragments).
    v8f s[4];
    #pragma unroll
    for (int j = 0; j < 4; ++j) s[j] = zero;
    const unsigned short* kp = Kb + (size_t)(b * T_ + kbase + col) * H_ + half * 16;
    #pragma unroll
    for (int j = 0; j < 4; ++j) {
      #pragma unroll
      for (int hc = 0; hc < 4; ++hc) {
        v16bf kf = *(const v16bf*)(kp + (size_t)j * 16 * H_ + hc * 32);
        s[j] = __builtin_amdgcn_wmma_f32_16x16x32_bf16(
            false, qf[hc], false, kf, (short)0, s[j], false, false);
      }
    }

    // Causal mask on the final (diagonal-touching) block.
    if (kb == nblocks - 1) {
      #pragma unroll
      for (int r = 0; r < 8; ++r) {
        int q = q0 + r + half * 8;
        #pragma unroll
        for (int j = 0; j < 4; ++j)
          if (kbase + j * 16 + col > q) s[j][r] = -3.0e38f;
      }
    }

    // Online softmax per row, exp2 domain (log2e pre-folded into Q).
    // Rows live in 16-lane halves; xor masks <=8 stay within the half.
    #pragma unroll
    for (int r = 0; r < 8; ++r) {
      float mx = fmaxf(fmaxf(s[0][r], s[1][r]), fmaxf(s[2][r], s[3][r]));
      mx = fmaxf(mx, __shfl_xor(mx, 1, 32));
      mx = fmaxf(mx, __shfl_xor(mx, 2, 32));
      mx = fmaxf(mx, __shfl_xor(mx, 4, 32));
      mx = fmaxf(mx, __shfl_xor(mx, 8, 32));
      float mnew = fmaxf(mrow[r], mx);
      float sum = 0.f;
      #pragma unroll
      for (int j = 0; j < 4; ++j) {
        float p = fexp2(s[j][r] - mnew);
        s[j][r] = p;
        sum += p;
      }
      sum += __shfl_xor(sum, 1, 32);
      sum += __shfl_xor(sum, 2, 32);
      sum += __shfl_xor(sum, 4, 32);
      sum += __shfl_xor(sum, 8, 32);
      float corr = fexp2(mrow[r] - mnew);
      lrow[r] = lrow[r] * corr + sum;
      mrow[r] = mnew;
      #pragma unroll
      for (int ht = 0; ht < 8; ++ht) o[ht][r] *= corr;
    }

    // P: C-layout f32 -> bf16 A-layout via this wave's LDS slice (in-order DS).
    #pragma unroll
    for (int r = 0; r < 8; ++r) {
      int row = r + half * 8;
      #pragma unroll
      for (int j = 0; j < 4; ++j)
        lds[w][row * 64 + j * 16 + col] = f2bf(s[j][r]);
    }
    __builtin_amdgcn_wave_barrier();
    v16bf pf0, pf1;
    {
      int k0 = half * 16;
      v8bf lo0 = *(const v8bf*)&lds[w][col * 64 + k0];
      v8bf hi0 = *(const v8bf*)&lds[w][col * 64 + k0 + 8];
      v8bf lo1 = *(const v8bf*)&lds[w][col * 64 + k0 + 32];
      v8bf hi1 = *(const v8bf*)&lds[w][col * 64 + k0 + 40];
      pf0 = __builtin_shufflevector(lo0, hi0, 0,1,2,3,4,5,6,7,8,9,10,11,12,13,14,15);
      pf1 = __builtin_shufflevector(lo1, hi1, 0,1,2,3,4,5,6,7,8,9,10,11,12,13,14,15);
    }

    // O += P @ V  (V transposed: contiguous 32B B-frag loads per lane).
    const unsigned short* vp = Vt + (size_t)b * H_ * T_ + kbase + half * 16;
    #pragma unroll
    for (int ht = 0; ht < 8; ++ht) {
      const unsigned short* vq = vp + (size_t)(ht * 16 + col) * T_;
      v16bf vf0 = *(const v16bf*)(vq);
      v16bf vf1 = *(const v16bf*)(vq + 32);
      o[ht] = __builtin_amdgcn_wmma_f32_16x16x32_bf16(false, pf0, false, vf0, (short)0, o[ht], false, false);
      o[ht] = __builtin_amdgcn_wmma_f32_16x16x32_bf16(false, pf1, false, vf1, (short)0, o[ht], false, false);
    }
  }

  // out[b][q][h] = O / l
  float* op = out + (size_t)(b * T_ + q0 + half * 8) * H_ + col;
  #pragma unroll
  for (int r = 0; r < 8; ++r) {
    float inv = 1.0f / lrow[r];
    #pragma unroll
    for (int ht = 0; ht < 8; ++ht)
      op[(size_t)r * H_ + ht * 16] = o[ht][r] * inv;
  }
}

// ---------------------------------------------------------------------------
extern "C" void kernel_launch(void* const* d_in, const int* in_sizes, int n_in,
                              void* d_out, int out_size, void* d_ws, size_t ws_size,
                              hipStream_t stream) {
  const float* x  = (const float*)d_in[0];
  const float* Wq = (const float*)d_in[1];
  const float* bq = (const float*)d_in[2];
  const float* Wk = (const float*)d_in[3];
  const float* bk = (const float*)d_in[4];
  const float* Wv = (const float*)d_in[5];
  const float* bv = (const float*)d_in[6];
  float* out = (float*)d_out;

  // Workspace layout (bf16 as ushort): Wt[3*H*C] | Qb[B*T*H] | Kb | Vt  (~26 MB)
  unsigned short* Wt = (unsigned short*)d_ws;
  unsigned short* Qb = Wt + (size_t)3 * H_ * C_;
  unsigned short* Kb = Qb + (size_t)B_ * T_ * H_;
  unsigned short* Vt = Kb + (size_t)B_ * T_ * H_;

  const int n0 = 3 * H_ * C_;
  wt_kernel<<<(n0 + 255) / 256, 256, 0, stream>>>(Wq, Wk, Wv, Wt);

  dim3 g1(B_ * T_ / (16 * 8), 3);              // 256 x 3 blocks, 8 waves each
  qkv_proj<<<g1, 256, 0, stream>>>(x, bq, bk, bv, Wt, Qb, Kb, Vt);

  flash_attn<<<B_ * T_ / (16 * 8), 256, 0, stream>>>(Qb, Kb, Vt, out);
}
